// BinaryDense_50861002719489
// MI455X (gfx1250) — matrix operand order repaired and checked by reading, not compiled, not run on hardware
//
#include <hip/hip_runtime.h>
#include <hip/hip_bf16.h>

// CDNA5 wave32 WMMA / TDM types
typedef __attribute__((ext_vector_type(16))) __bf16       v16bf;
typedef __attribute__((ext_vector_type(8)))  float        v8f;
typedef __attribute__((ext_vector_type(4)))  unsigned int v4u;
typedef unsigned int u32x4 __attribute__((ext_vector_type(4)));
typedef int          i32x4 __attribute__((ext_vector_type(4)));
typedef int          i32x8 __attribute__((ext_vector_type(8)));

__device__ __forceinline__ unsigned short f32_bf16_rne(float f) {
    union { float f; unsigned u; } v; v.f = f;
    unsigned r = 0x7FFFu + ((v.u >> 16) & 1u);
    return (unsigned short)((v.u + r) >> 16);
}

union Frag {            // 32 bytes: 16 bf16 = 8 VGPRs
    v16bf v;
    v4u   u[2];
};

// ---------------------------------------------------------------------------
// Pass 1: X f32 -> bf16 (RNE)
// ---------------------------------------------------------------------------
__global__ __launch_bounds__(256)
void convert_x_bf16(const float* __restrict__ X, unsigned short* __restrict__ Xb,
                    long long n4) {
    long long i = (long long)blockIdx.x * blockDim.x + threadIdx.x;
    long long stride = (long long)gridDim.x * blockDim.x;
    for (; i < n4; i += stride) {
        float4 f = ((const float4*)X)[i];
        uint2 pk;
        pk.x = (unsigned)f32_bf16_rne(f.x) | ((unsigned)f32_bf16_rne(f.y) << 16);
        pk.y = (unsigned)f32_bf16_rne(f.z) | ((unsigned)f32_bf16_rne(f.w) << 16);
        ((uint2*)Xb)[i] = pk;
    }
}

// ---------------------------------------------------------------------------
// Pass 2: W f32 -> binarized ±1 bf16 (w>0 -> +1 else -1)
// ---------------------------------------------------------------------------
__global__ __launch_bounds__(256)
void convert_w_bin(const float* __restrict__ W, unsigned short* __restrict__ Wb,
                   long long n4) {
    long long i = (long long)blockIdx.x * blockDim.x + threadIdx.x;
    long long stride = (long long)gridDim.x * blockDim.x;
    for (; i < n4; i += stride) {
        float4 f = ((const float4*)W)[i];
        unsigned s0 = (f.x > 0.f) ? 0x3F80u : 0xBF80u;
        unsigned s1 = (f.y > 0.f) ? 0x3F80u : 0xBF80u;
        unsigned s2 = (f.z > 0.f) ? 0x3F80u : 0xBF80u;
        unsigned s3 = (f.w > 0.f) ? 0x3F80u : 0xBF80u;
        uint2 pk; pk.x = s0 | (s1 << 16); pk.y = s2 | (s3 << 16);
        ((uint2*)Wb)[i] = pk;
    }
}

// ---------------------------------------------------------------------------
// TDM: issue a 2D tensor_load_to_lds per ISA D# layout (§8.3-8.5).
//   data_size = 4B units. Padding: every 64 dwords (pad_interval=5) insert
//   4 dwords (pad_amount=3) -> rows of 128 bf16 padded to 136 in LDS.
// ---------------------------------------------------------------------------
#if defined(__gfx1250__) && __has_builtin(__builtin_amdgcn_tensor_load_to_lds)
#define USE_TDM 1
__device__ __forceinline__ void tdm_load_2d(unsigned lds_addr, const void* gptr,
                                            unsigned tensor_d0, unsigned tensor_d1,
                                            unsigned long long stride0,
                                            unsigned tile_d0, unsigned tile_d1) {
    unsigned long long ga = (unsigned long long)(size_t)gptr;
    u32x4 g0;
    g0[0] = 1u;                                              // count=1, user mode
    g0[1] = lds_addr;                                        // lds_addr[63:32]
    g0[2] = (unsigned)(ga & 0xFFFFFFFFull);                  // global_addr[95:64]
    g0[3] = (unsigned)((ga >> 32) & 0x01FFFFFFull) | (2u << 30); // addr[120:96], type=2

    i32x8 g1;
    g1[0] = (int)((2u << 16)        // data_size = 4B
                | (1u << 20)        // pad_enable
                | (5u << 22)        // pad_interval: 64 dwords
                | (3u << 25));      // pad_amount: 4 dwords
    g1[1] = (int)((tensor_d0 & 0xFFFFu) << 16);              // tensor_dim0[15:0]
    g1[2] = (int)((tensor_d0 >> 16) | ((tensor_d1 & 0xFFFFu) << 16));
    g1[3] = (int)((tensor_d1 >> 16) | (tile_d0 << 16));      // tile_dim0
    g1[4] = (int)(tile_d1);                                  // tile_dim1, tile_dim2=0
    g1[5] = (int)(unsigned)(stride0 & 0xFFFFFFFFull);        // dim0_stride[31:0]
    g1[6] = (int)(unsigned)((stride0 >> 32) & 0xFFFFull);    // dim0_stride[47:32]
    g1[7] = 0;

    i32x4 gz = {0, 0, 0, 0};
#if __clang_major__ >= 23
    i32x8 gz8 = {0, 0, 0, 0, 0, 0, 0, 0};
    __builtin_amdgcn_tensor_load_to_lds(g0, g1, gz, gz, gz8, 0);
#else
    __builtin_amdgcn_tensor_load_to_lds(g0, g1, gz, gz, 0);
#endif
}
#endif

// ---------------------------------------------------------------------------
// Pass 3: bf16 GEMM  O[M,N] = Xb[M,K] * Wb[N,K]^T
//   block tile 128x64, 8 waves (4x2), wave tile 32x32 (2x2 WMMA accumulators)
//   TILE_K=128 -> 16 WMMAs/wave per barrier; double-buffered LDS fed by TDM
// ---------------------------------------------------------------------------
#define G_TILE_M 128
#define G_TILE_N 64
#define G_TILE_K 128
#define G_LDSK   (G_TILE_K + 8)   // 136 elems = 272B row stride, 16B aligned

__global__ __launch_bounds__(256, 2)
void gemm_bf16_wmma(const unsigned short* __restrict__ Xb,  // [M,K] bf16
                    const unsigned short* __restrict__ Wb,  // [N,K] bf16 ±1
                    float* __restrict__ O,                  // [M,N] f32
                    int M, int N, int K) {
    __shared__ __align__(16) unsigned short lds_x[2][G_TILE_M][G_LDSK];
    __shared__ __align__(16) unsigned short lds_w[2][G_TILE_N][G_LDSK];

    const int tid    = threadIdx.x;
    const int wave   = tid >> 5;
    const int lane   = tid & 31;
    const int waveM  = wave & 3;
    const int waveN  = wave >> 2;
    const int lane15 = lane & 15;
    const int laneHi = lane >> 4;

    const int blockN0 = blockIdx.x * G_TILE_N;
    const int blockM0 = blockIdx.y * G_TILE_M;

    v8f acc[2][2];
    acc[0][0] = {}; acc[0][1] = {}; acc[1][0] = {}; acc[1][1] = {};

#ifdef USE_TDM
    // --- TDM-pipelined path: wave 0 drives the Tensor Data Mover -------------
    const unsigned td0 = (unsigned)(K / 2);                  // dwords per row
    if (wave == 0) {     // prologue: tile 0 into buffer 0
        tdm_load_2d((unsigned)(size_t)&lds_x[0][0][0],
                    Xb + (size_t)blockM0 * K, td0, (unsigned)M,
                    (unsigned long long)(K / 2), G_TILE_K / 2, G_TILE_M);
        tdm_load_2d((unsigned)(size_t)&lds_w[0][0][0],
                    Wb + (size_t)blockN0 * K, td0, (unsigned)N,
                    (unsigned long long)(K / 2), G_TILE_K / 2, G_TILE_N);
        __builtin_amdgcn_s_wait_tensorcnt(0);
    }
    __syncthreads();

    int buf = 0;
    for (int k0 = 0; k0 < K; k0 += G_TILE_K) {
        const int knext = k0 + G_TILE_K;
        if (wave == 0 && knext < K) {   // prefetch next tile into other buffer
            tdm_load_2d((unsigned)(size_t)&lds_x[buf ^ 1][0][0],
                        Xb + (size_t)blockM0 * K + knext, td0, (unsigned)M,
                        (unsigned long long)(K / 2), G_TILE_K / 2, G_TILE_M);
            tdm_load_2d((unsigned)(size_t)&lds_w[buf ^ 1][0][0],
                        Wb + (size_t)blockN0 * K + knext, td0, (unsigned)N,
                        (unsigned long long)(K / 2), G_TILE_K / 2, G_TILE_N);
        }

        #pragma unroll
        for (int kk = 0; kk < G_TILE_K; kk += 32) {
            Frag a[2], b[2];
            const int akoff = kk + laneHi * 8;
            #pragma unroll
            for (int i = 0; i < 2; ++i) {
                const unsigned short* p = &lds_x[buf][waveM * 32 + i * 16 + lane15][akoff];
                a[i].u[0] = *(const v4u*)(p);
                a[i].u[1] = *(const v4u*)(p + 16);
            }
            const int bkoff = kk + laneHi * 8;
            #pragma unroll
            for (int j = 0; j < 2; ++j) {
                const unsigned short* p = &lds_w[buf][waveN * 32 + j * 16 + lane15][bkoff];
                b[j].u[0] = *(const v4u*)(p);
                b[j].u[1] = *(const v4u*)(p + 16);
            }
            #pragma unroll
            for (int i = 0; i < 2; ++i)
                #pragma unroll
                for (int j = 0; j < 2; ++j)
                    acc[i][j] = __builtin_amdgcn_wmma_f32_16x16x32_bf16(
                        false, a[i].v, false, b[j].v,
                        (short)0, acc[i][j], false, false);
        }

        if (wave == 0) __builtin_amdgcn_s_wait_tensorcnt(0);
        __syncthreads();
        buf ^= 1;
    }
#else
    // --- Generic path (host pass / toolchains without the TDM builtin) ------
    for (int k0 = 0; k0 < K; k0 += G_TILE_K) {
        #pragma unroll
        for (int it = 0; it < 8; ++it) {
            int idx = tid + it * 256;
            int row = idx >> 4;
            int q   = idx & 15;
            v4u d = *(const v4u*)(Xb + (size_t)(blockM0 + row) * K + k0 + q * 8);
            *(v4u*)&lds_x[0][row][q * 8] = d;
        }
        #pragma unroll
        for (int it = 0; it < 4; ++it) {
            int idx = tid + it * 256;
            int row = idx >> 4;
            int q   = idx & 15;
            v4u d = *(const v4u*)(Wb + (size_t)(blockN0 + row) * K + k0 + q * 8);
            *(v4u*)&lds_w[0][row][q * 8] = d;
        }
        __syncthreads();

        #pragma unroll
        for (int kk = 0; kk < G_TILE_K; kk += 32) {
            Frag a[2], b[2];
            const int akoff = kk + laneHi * 8;
            #pragma unroll
            for (int i = 0; i < 2; ++i) {
                const unsigned short* p = &lds_x[0][waveM * 32 + i * 16 + lane15][akoff];
                a[i].u[0] = *(const v4u*)(p);
                a[i].u[1] = *(const v4u*)(p + 16);
            }
            const int bkoff = kk + laneHi * 8;
            #pragma unroll
            for (int j = 0; j < 2; ++j) {
                const unsigned short* p = &lds_w[0][waveN * 32 + j * 16 + lane15][bkoff];
                b[j].u[0] = *(const v4u*)(p);
                b[j].u[1] = *(const v4u*)(p + 16);
            }
            #pragma unroll
            for (int i = 0; i < 2; ++i)
                #pragma unroll
                for (int j = 0; j < 2; ++j)
                    acc[i][j] = __builtin_amdgcn_wmma_f32_16x16x32_bf16(
                        false, a[i].v, false, b[j].v,
                        (short)0, acc[i][j], false, false);
        }
        __syncthreads();
    }
#endif

    const int mb = blockM0 + waveM * 32;
    const int nb = blockN0 + waveN * 32;
    #pragma unroll
    for (int i = 0; i < 2; ++i)
        #pragma unroll
        for (int j = 0; j < 2; ++j) {
            const int n = nb + j * 16 + lane15;
            #pragma unroll
            for (int r = 0; r < 8; ++r) {
                const int m = mb + i * 16 + r + laneHi * 8;
                O[(size_t)m * N + n] = acc[i][j][r];
            }
        }
}

// ---------------------------------------------------------------------------
// Fallback: fused single-pass kernel (only if ws_size too small)
// ---------------------------------------------------------------------------
#define TILE_M 128
#define TILE_N 64
#define TILE_K 64
#define LDSK   (TILE_K + 8)

__global__ __launch_bounds__(256, 2)
void binary_dense_fused(const float* __restrict__ X, const float* __restrict__ W,
                        float* __restrict__ O, int M, int N, int K) {
    __shared__ __align__(16) unsigned short lds_x[TILE_M][LDSK];
    __shared__ __align__(16) unsigned short lds_w[TILE_N][LDSK];

    const int tid    = threadIdx.x;
    const int wave   = tid >> 5;
    const int lane   = tid & 31;
    const int waveM  = wave & 3;
    const int waveN  = wave >> 2;
    const int lane15 = lane & 15;
    const int laneHi = lane >> 4;

    const int blockN0 = blockIdx.x * TILE_N;
    const int blockM0 = blockIdx.y * TILE_M;

    v8f acc[2][2];
    acc[0][0] = {}; acc[0][1] = {}; acc[1][0] = {}; acc[1][1] = {};

    for (int k0 = 0; k0 < K; k0 += TILE_K) {
        #pragma unroll
        for (int it = 0; it < 8; ++it) {
            int idx = tid + it * 256;
            int row = idx >> 4;
            int q   = idx & 15;
            float4 f = ((const float4*)(X + (size_t)(blockM0 + row) * K + k0))[q];
            uint2 pk;
            pk.x = (unsigned)f32_bf16_rne(f.x) | ((unsigned)f32_bf16_rne(f.y) << 16);
            pk.y = (unsigned)f32_bf16_rne(f.z) | ((unsigned)f32_bf16_rne(f.w) << 16);
            *(uint2*)&lds_x[row][q * 4] = pk;
        }
        #pragma unroll
        for (int it = 0; it < 4; ++it) {
            int idx = tid + it * 256;
            int row = idx >> 4;
            int q   = idx & 15;
            float4 f = ((const float4*)(W + (size_t)(blockN0 + row) * K + k0))[q];
            unsigned s0 = (f.x > 0.f) ? 0x3F80u : 0xBF80u;
            unsigned s1 = (f.y > 0.f) ? 0x3F80u : 0xBF80u;
            unsigned s2 = (f.z > 0.f) ? 0x3F80u : 0xBF80u;
            unsigned s3 = (f.w > 0.f) ? 0x3F80u : 0xBF80u;
            uint2 pk; pk.x = s0 | (s1 << 16); pk.y = s2 | (s3 << 16);
            *(uint2*)&lds_w[row][q * 4] = pk;
        }
        __syncthreads();

        #pragma unroll
        for (int kk = 0; kk < TILE_K; kk += 32) {
            Frag a[2], b[2];
            const int akoff = kk + laneHi * 8;
            #pragma unroll
            for (int i = 0; i < 2; ++i) {
                const unsigned short* p = &lds_x[waveM * 32 + i * 16 + lane15][akoff];
                a[i].u[0] = *(const v4u*)(p);
                a[i].u[1] = *(const v4u*)(p + 16);
            }
            const int bkoff = kk + laneHi * 8;
            #pragma unroll
            for (int j = 0; j < 2; ++j) {
                const unsigned short* p = &lds_w[waveN * 32 + j * 16 + lane15][bkoff];
                b[j].u[0] = *(const v4u*)(p);
                b[j].u[1] = *(const v4u*)(p + 16);
            }
            #pragma unroll
            for (int i = 0; i < 2; ++i)
                #pragma unroll
                for (int j = 0; j < 2; ++j)
                    acc[i][j] = __builtin_amdgcn_wmma_f32_16x16x32_bf16(
                        false, a[i].v, false, b[j].v,
                        (short)0, acc[i][j], false, false);
        }
        __syncthreads();
    }

    const int mb = blockM0 + waveM * 32;
    const int nb = blockN0 + waveN * 32;
    #pragma unroll
    for (int i = 0; i < 2; ++i)
        #pragma unroll
        for (int j = 0; j < 2; ++j) {
            const int n = nb + j * 16 + lane15;
            #pragma unroll
            for (int r = 0; r < 8; ++r) {
                const int m = mb + i * 16 + r + laneHi * 8;
                O[(size_t)m * N + n] = acc[i][j][r];
            }
        }
}

// ---------------------------------------------------------------------------
extern "C" void kernel_launch(void* const* d_in, const int* in_sizes, int n_in,
                              void* d_out, int out_size, void* d_ws, size_t ws_size,
                              hipStream_t stream) {
    const float* x = (const float*)d_in[0];   // [M, K] f32
    const float* w = (const float*)d_in[1];   // [N, K] f32
    float* out = (float*)d_out;               // [M, N] f32

    const int K = 4096;
    const int N = in_sizes[1] / K;            // 4096
    const int M = in_sizes[0] / K;            // 8192

    const size_t needX = (size_t)M * K * sizeof(unsigned short);
    const size_t needW = (size_t)N * K * sizeof(unsigned short);

    if (ws_size >= needX + needW) {
        unsigned short* Xb = (unsigned short*)d_ws;
        unsigned short* Wb = (unsigned short*)((char*)d_ws + needX);

        long long nx4 = (long long)M * K / 4;
        long long nw4 = (long long)N * K / 4;
        convert_x_bf16<<<dim3(4096), dim3(256), 0, stream>>>(x, Xb, nx4);
        convert_w_bin<<<dim3(2048), dim3(256), 0, stream>>>(w, Wb, nw4);

        dim3 grid(N / G_TILE_N, M / G_TILE_M);   // (64, 64)
        gemm_bf16_wmma<<<grid, dim3(256), 0, stream>>>(Xb, Wb, out, M, N, K);
    } else {
        dim3 grid(N / TILE_N, M / TILE_M);
        binary_dense_fused<<<grid, dim3(256), 0, stream>>>(x, w, out, M, N, K);
    }
}
